// GNN_2_60894046322998
// MI455X (gfx1250) — compile-verified, matrix operand order checked by reading, not run on hardware
//
#include <hip/hip_runtime.h>
#include <hip/hip_bf16.h>
#include <math.h>

// ---------------- problem constants (match reference) ----------------
#define N_NODES 50000
#define N_EDGES 800000
#define E2      (N_EDGES + N_NODES)   // edges + self loops
#define HEADS   4
#define CH      64
#define HC      256                    // HEADS*CH == F_IN
#define NGRAPH  128
#define DENSE   64
#define NCLS    2
#define NEG_SLOPE 0.2f
#define BN_EPS    1e-5f

typedef __bf16 bf16;
typedef bf16  v16bf __attribute__((ext_vector_type(16)));
typedef float v8f   __attribute__((ext_vector_type(8)));

// ---------------- small utility kernels ----------------
__global__ void k_fill_f32(float* p, float v, int n) {
    int i = blockIdx.x * blockDim.x + threadIdx.x;
    if (i < n) p[i] = v;
}
__global__ void k_fill_u32(unsigned* p, unsigned v, int n) {
    int i = blockIdx.x * blockDim.x + threadIdx.x;
    if (i < n) p[i] = v;
}
__global__ void k_cvt_bf16(const float* __restrict__ in, bf16* __restrict__ out, int n) {
    int i = blockIdx.x * blockDim.x + threadIdx.x;
    if (i < n) out[i] = (bf16)in[i];
}
// W [HC,HC] row-major (k,n) -> WT bf16 [n,k] so B-fragment loads are contiguous
__global__ void k_wt_cvt(const float* __restrict__ W, bf16* __restrict__ WT) {
    int i = blockIdx.x * blockDim.x + threadIdx.x;
    if (i < HC * HC) {
        int n = i / HC, k = i % HC;
        WT[n * HC + k] = (bf16)W[k * HC + n];
    }
}

// ---------------- WMMA GEMM: C[N_NODES,HC] = A[N_NODES,HC] * W[HC,HC] ----------------
// A is bf16 row-major, BT is W^T bf16 row-major (row = output channel n).
// One wave per 16x16 output tile; block = 16 waves covering all 16 n-tiles of one m-tile.
// 16-bit A-fragment layout (ISA 7.12.2): lane 0-15 -> M=lane, halves select K groups,
// dword i of the fragment holds K pair: i<4 -> K=half*8+2i, else K=16+half*8+2(i-4).
__global__ __launch_bounds__(512) void k_gemm_wmma(const bf16* __restrict__ A,
                                                   const bf16* __restrict__ BT,
                                                   float* __restrict__ Cout) {
    const int lane  = threadIdx.x & 31;
    const int wave  = threadIdx.x >> 5;          // n-tile 0..15
    const int mtile = blockIdx.x;                // N_NODES/16 tiles (50000 = 16*3125)
    const int half  = lane >> 4;
    const int mrow  = mtile * 16 + (lane & 15);
    const int ncol  = wave * 16 + (lane & 15);

    const unsigned* arow = (const unsigned*)(A  + (size_t)mrow * HC);  // bf16 pairs
    const unsigned* brow = (const unsigned*)(BT + (size_t)ncol * HC);

    union Frag { v16bf v; unsigned u[8]; };
    v8f acc = {};
#pragma unroll
    for (int ks = 0; ks < 8; ++ks) {             // K = 256 in steps of 32
        Frag a, b;
        const int kd = ks * 16;                  // dword base (= K/2)
#pragma unroll
        for (int i = 0; i < 8; ++i) {
            int di = kd + ((i < 4) ? (half * 4 + i) : (8 + half * 4 + (i - 4)));
            a.u[i] = arow[di];
            b.u[i] = brow[di];
        }
        acc = __builtin_amdgcn_wmma_f32_16x16x32_bf16(
                  false, a.v, false, b.v, (short)0, acc, false, false);
    }
    // C layout: VGPR r -> M = r + 8*half, N = lane&15 within tile
#pragma unroll
    for (int r = 0; r < 8; ++r) {
        int m = mtile * 16 + r + half * 8;
        Cout[(size_t)m * HC + ncol] = acc[r];
    }
}

// ---------------- attention coefficients per node/head ----------------
__global__ void k_attn_coef(const float* __restrict__ h, const float* __restrict__ asrc,
                            const float* __restrict__ adst, float* es, float* ed) {
    int i = blockIdx.x * blockDim.x + threadIdx.x;
    if (i >= N_NODES * HEADS) return;
    int n = i / HEADS, hh = i % HEADS;
    const float* hp = h + (size_t)n * HC + hh * CH;
    const float* as = asrc + hh * CH;
    const float* ad = adst + hh * CH;
    float s0 = 0.f, s1 = 0.f;
#pragma unroll 4
    for (int c = 0; c < CH; ++c) { float v = hp[c]; s0 += v * as[c]; s1 += v * ad[c]; }
    es[i] = s0; ed[i] = s1;
}

__device__ __forceinline__ void edge_sd(const int* ei, int e, int& s, int& d) {
    if (e < N_EDGES) { s = ei[e]; d = ei[N_EDGES + e]; }
    else             { s = d = e - N_EDGES; }          // self loop
}
// order-preserving float<->uint mapping for atomic max
__device__ __forceinline__ unsigned f2ord(float f) {
    unsigned u = __float_as_uint(f);
    return (u & 0x80000000u) ? ~u : (u | 0x80000000u);
}
__device__ __forceinline__ float ord2f(unsigned x) {
    return (x & 0x80000000u) ? __uint_as_float(x ^ 0x80000000u) : __uint_as_float(~x);
}

// pass 1: e = leakyrelu(es[src]+ed[dst]); segment max into dst
__global__ void k_edge_logits(const int* __restrict__ ei, const float* __restrict__ es,
                              const float* __restrict__ ed, float* __restrict__ ebuf,
                              unsigned* __restrict__ mmax) {
    int i = blockIdx.x * blockDim.x + threadIdx.x;
    if (i >= E2 * HEADS) return;
    int e = i / HEADS, hh = i % HEADS;
    int s, d; edge_sd(ei, e, s, d);
    float v = es[s * HEADS + hh] + ed[d * HEADS + hh];
    v = (v >= 0.f) ? v : NEG_SLOPE * v;
    ebuf[i] = v;
    atomicMax(&mmax[d * HEADS + hh], f2ord(v));
}
// pass 2: ex = exp(e - max); segment sum into dst
__global__ void k_edge_exp(const int* __restrict__ ei, float* __restrict__ ebuf,
                           const unsigned* __restrict__ mmax, float* __restrict__ ssum) {
    int i = blockIdx.x * blockDim.x + threadIdx.x;
    if (i >= E2 * HEADS) return;
    int e = i / HEADS, hh = i % HEADS;
    int s, d; edge_sd(ei, e, s, d);
    float ex = __expf(ebuf[i] - ord2f(mmax[d * HEADS + hh]));
    ebuf[i] = ex;
    atomicAdd(&ssum[d * HEADS + hh], ex);
}
// agg initialized with GAT bias
__global__ void k_init_bias(float* agg, const float* __restrict__ bias, int total) {
    int i = blockIdx.x * blockDim.x + threadIdx.x;
    if (i < total) agg[i] = bias[i % HC];
}
// pass 3: one wave per edge, lane l owns channels [8l,8l+8); alpha shared per head
__global__ __launch_bounds__(256) void k_edge_agg(const int* __restrict__ ei,
                                                  const float* __restrict__ ebuf,
                                                  const float* __restrict__ ssum,
                                                  const float* __restrict__ h,
                                                  float* __restrict__ agg) {
    int lane = threadIdx.x & 31;
    int e = blockIdx.x * (blockDim.x >> 5) + (threadIdx.x >> 5);
    if (e >= E2) return;
    int s, d; edge_sd(ei, e, s, d);
    int hh = lane >> 3;                                  // 8 lanes per head
    float alpha = ebuf[e * HEADS + hh] / (ssum[d * HEADS + hh] + 1e-16f);
    int c0 = lane * 8;
    const float* hs = h + (size_t)s * HC;
    float* ao = agg + (size_t)d * HC;
#pragma unroll
    for (int j = 0; j < 8; ++j) atomicAdd(&ao[c0 + j], hs[c0 + j] * alpha);
}

__global__ void k_relu(const float* __restrict__ in, float* __restrict__ out, int n) {
    int i = blockIdx.x * blockDim.x + threadIdx.x;
    if (i < n) out[i] = fmaxf(in[i], 0.f);
}

// BatchNorm over nodes: biased batch variance. One block per channel.
__global__ __launch_bounds__(256) void k_bn_stats(const float* __restrict__ x,
                                                  float* mu, float* var, int rows) {
    int ch = blockIdx.x;
    __shared__ float s1[256], s2[256];
    float a = 0.f, b = 0.f;
    for (int n = threadIdx.x; n < rows; n += 256) {
        float v = x[(size_t)n * HC + ch];
        a += v; b += v * v;
    }
    s1[threadIdx.x] = a; s2[threadIdx.x] = b; __syncthreads();
    for (int st = 128; st > 0; st >>= 1) {
        if (threadIdx.x < st) { s1[threadIdx.x] += s1[threadIdx.x + st];
                                s2[threadIdx.x] += s2[threadIdx.x + st]; }
        __syncthreads();
    }
    if (threadIdx.x == 0) {
        float m = s1[0] / rows;
        mu[ch] = m; var[ch] = s2[0] / rows - m * m;
    }
}
__global__ void k_bn_relu(const float* __restrict__ x, const float* __restrict__ mu,
                          const float* __restrict__ var, const float* __restrict__ g,
                          const float* __restrict__ b, float* __restrict__ out, int total) {
    int i = blockIdx.x * blockDim.x + threadIdx.x;
    if (i >= total) return;
    int ch = i % HC;
    float y = (x[i] - mu[ch]) * rsqrtf(var[ch] + BN_EPS) * g[ch] + b[ch];
    out[i] = fmaxf(y, 0.f);
}

// ---------------- pooling ----------------
__global__ void k_count(const int* __restrict__ batch, float* cnt) {
    int n = blockIdx.x * blockDim.x + threadIdx.x;
    if (n < N_NODES) atomicAdd(&cnt[batch[n]], 1.f);
}
__global__ void k_pool(const float* __restrict__ act, const int* __restrict__ batch,
                       float* __restrict__ ps) {
    int i = blockIdx.x * blockDim.x + threadIdx.x;
    if (i < N_NODES * HC) {
        int n = i / HC, ch = i % HC;
        atomicAdd(&ps[batch[n] * HC + ch], act[i]);
    }
}
__global__ void k_pool_fin(const float* __restrict__ ps, const float* __restrict__ cnt,
                           float* __restrict__ outPooled) {
    int i = blockIdx.x * blockDim.x + threadIdx.x;
    if (i < NGRAPH * HC) outPooled[i] = ps[i] / fmaxf(cnt[i / HC], 1.f);
}
__global__ void k_bn1_stats(const float* __restrict__ pooled, float* mu, float* var) {
    int ch = threadIdx.x;   // 256 threads
    float a = 0.f, b = 0.f;
    for (int g = 0; g < NGRAPH; ++g) { float v = pooled[g * HC + ch]; a += v; b += v * v; }
    float m = a / NGRAPH;
    mu[ch] = m; var[ch] = b / NGRAPH - m * m;
}

// ---------------- MLP head, one block per graph ----------------
__global__ __launch_bounds__(64) void k_head(const float* __restrict__ pooled,
        const float* __restrict__ mu, const float* __restrict__ var,
        const float* __restrict__ g1, const float* __restrict__ b1,
        const float* __restrict__ W1h, const float* __restrict__ b1h,
        const float* __restrict__ W2h, const float* __restrict__ b2h,
        const float* __restrict__ W3h, const float* __restrict__ b3h,
        float* __restrict__ out) {
    __shared__ float z[HC], z2[DENSE], z3[DENSE], o[NCLS];
    int g = blockIdx.x, t = threadIdx.x;
    for (int ch = t; ch < HC; ch += 64) {
        float v = pooled[g * HC + ch];
        z[ch] = (v - mu[ch]) * rsqrtf(var[ch] + BN_EPS) * g1[ch] + b1[ch];
    }
    __syncthreads();
    float acc = b1h[t];
    for (int k = 0; k < HC; ++k) acc += z[k] * W1h[k * DENSE + t];
    z2[t] = fmaxf(acc, 0.f);
    __syncthreads();
    acc = b2h[t];
    for (int k = 0; k < DENSE; ++k) acc += z2[k] * W2h[k * DENSE + t];
    z3[t] = fmaxf(acc, 0.f);
    __syncthreads();
    if (t < NCLS) {
        acc = b3h[t];
        for (int k = 0; k < DENSE; ++k) acc += z3[k] * W3h[k * NCLS + t];
        o[t] = fmaxf(acc, 0.f);
    }
    __syncthreads();
    if (t == 0) {
        float m = fmaxf(o[0], o[1]);
        float e0 = __expf(o[0] - m), e1 = __expf(o[1] - m);
        float s = e0 + e1;
        out[g * NCLS + 0] = e0 / s;
        out[g * NCLS + 1] = e1 / s;
    }
}

// ---------------- host orchestration ----------------
extern "C" void kernel_launch(void* const* d_in, const int* in_sizes, int n_in,
                              void* d_out, int out_size, void* d_ws, size_t ws_size,
                              hipStream_t stream) {
    (void)in_sizes; (void)n_in; (void)out_size; (void)ws_size;
    const float* x     = (const float*)d_in[0];
    const int*   ei    = (const int*)d_in[1];
    const int*   batch = (const int*)d_in[2];
    const float* W1    = (const float*)d_in[3];
    const float* as1   = (const float*)d_in[4];
    const float* ad1   = (const float*)d_in[5];
    const float* b1    = (const float*)d_in[6];
    const float* Wl    = (const float*)d_in[7];
    const float* asl   = (const float*)d_in[8];
    const float* adl   = (const float*)d_in[9];
    const float* bl    = (const float*)d_in[10];
    const float* bng   = (const float*)d_in[11];
    const float* bnb   = (const float*)d_in[12];
    const float* bn1g  = (const float*)d_in[13];
    const float* bn1b  = (const float*)d_in[14];
    const float* lin1W = (const float*)d_in[15];
    const float* lin1b = (const float*)d_in[16];
    const float* lin2W = (const float*)d_in[17];
    const float* lin2b = (const float*)d_in[18];
    const float* lin3W = (const float*)d_in[19];
    const float* lin3b = (const float*)d_in[20];

    float* out       = (float*)d_out;
    float* outPooled = out + NGRAPH * NCLS;   // tuple order: (softmax out, pooled)

    // deterministic workspace carve-out (256B aligned)
    char* ws = (char*)d_ws;
    size_t off = 0;
    auto carve = [&](size_t bytes) -> char* {
        char* p = ws + off;
        off = (off + bytes + 255) & ~(size_t)255;
        return p;
    };
    bf16* wt0 = (bf16*)carve((size_t)HC * HC * sizeof(bf16));
    bf16* wt1 = (bf16*)carve((size_t)HC * HC * sizeof(bf16));
    bf16* wt2 = (bf16*)carve((size_t)HC * HC * sizeof(bf16));
    bf16* xb  = (bf16*)carve((size_t)N_NODES * HC * sizeof(bf16));
    float* hlin = (float*)carve((size_t)N_NODES * HC * sizeof(float)); // GEMM out / activation
    float* agg  = (float*)carve((size_t)N_NODES * HC * sizeof(float));
    float* es   = (float*)carve((size_t)N_NODES * HEADS * sizeof(float));
    float* ed   = (float*)carve((size_t)N_NODES * HEADS * sizeof(float));
    unsigned* mmax = (unsigned*)carve((size_t)N_NODES * HEADS * sizeof(unsigned));
    float* ssum = (float*)carve((size_t)N_NODES * HEADS * sizeof(float));
    float* ebuf = (float*)carve((size_t)E2 * HEADS * sizeof(float));
    float* psum = (float*)carve((size_t)NGRAPH * HC * sizeof(float));
    float* cnt  = (float*)carve((size_t)NGRAPH * sizeof(float));
    float* mu   = (float*)carve(HC * sizeof(float));
    float* var  = (float*)carve(HC * sizeof(float));
    float* mu1  = (float*)carve(HC * sizeof(float));
    float* var1 = (float*)carve(HC * sizeof(float));

    const int T = 256;
    const int welems = HC * HC;
    k_wt_cvt<<<(welems + T - 1) / T, T, 0, stream>>>(W1, wt0);
    k_wt_cvt<<<(welems + T - 1) / T, T, 0, stream>>>(Wl, wt1);
    k_wt_cvt<<<(welems + T - 1) / T, T, 0, stream>>>(Wl + welems, wt2);

    bf16* wt[3] = {wt0, wt1, wt2};
    const float* aS[3]   = {as1, asl, asl + HEADS * CH};
    const float* aD[3]   = {ad1, adl, adl + HEADS * CH};
    const float* bias[3] = {b1, bl, bl + HC};

    const int total = N_NODES * HC;
    const int nhTot = N_NODES * HEADS;
    const int eTot  = E2 * HEADS;

    for (int l = 0; l < 3; ++l) {
        const float* src = (l == 0) ? x : hlin;
        k_cvt_bf16<<<(total + T - 1) / T, T, 0, stream>>>(src, xb, total);
        k_gemm_wmma<<<N_NODES / 16, 512, 0, stream>>>(xb, wt[l], hlin);
        k_attn_coef<<<(nhTot + T - 1) / T, T, 0, stream>>>(hlin, aS[l], aD[l], es, ed);
        k_fill_u32<<<(nhTot + T - 1) / T, T, 0, stream>>>(mmax, 0u, nhTot);
        k_fill_f32<<<(nhTot + T - 1) / T, T, 0, stream>>>(ssum, 0.f, nhTot);
        k_edge_logits<<<(eTot + T - 1) / T, T, 0, stream>>>(ei, es, ed, ebuf, mmax);
        k_edge_exp<<<(eTot + T - 1) / T, T, 0, stream>>>(ei, ebuf, mmax, ssum);
        k_init_bias<<<(total + T - 1) / T, T, 0, stream>>>(agg, bias[l], total);
        k_edge_agg<<<(E2 + 7) / 8, T, 0, stream>>>(ei, ebuf, ssum, hlin, agg);
        if (l == 0) {
            k_relu<<<(total + T - 1) / T, T, 0, stream>>>(agg, hlin, total);
        } else {
            k_bn_stats<<<HC, T, 0, stream>>>(agg, mu, var, N_NODES);
            k_bn_relu<<<(total + T - 1) / T, T, 0, stream>>>(
                agg, mu, var, bng + (l - 1) * HC, bnb + (l - 1) * HC, hlin, total);
        }
    }

    k_fill_f32<<<(NGRAPH * HC + T - 1) / T, T, 0, stream>>>(psum, 0.f, NGRAPH * HC);
    k_fill_f32<<<1, NGRAPH, 0, stream>>>(cnt, 0.f, NGRAPH);
    k_count<<<(N_NODES + T - 1) / T, T, 0, stream>>>(batch, cnt);
    k_pool<<<(total + T - 1) / T, T, 0, stream>>>(hlin, batch, psum);
    k_pool_fin<<<(NGRAPH * HC + T - 1) / T, T, 0, stream>>>(psum, cnt, outPooled);
    k_bn1_stats<<<1, HC, 0, stream>>>(outPooled, mu1, var1);
    k_head<<<NGRAPH, 64, 0, stream>>>(outPooled, mu1, var1, bn1g, bn1b,
                                      lin1W, lin1b, lin2W, lin2b, lin3W, lin3b, out);
}
